// GraphUNet_867583393848
// MI455X (gfx1250) — matrix operand order repaired and checked by reading, not compiled, not run on hardware
//
#include <hip/hip_runtime.h>
#include <hip/hip_bf16.h>
#include <cstdint>
#include <cstddef>

// ---------------- problem constants (match reference) ----------------
#define B_GR   64
#define NPER   1024
#define E_TOT  (B_GR * NPER * 16)     // 1,048,576 edges
#define K1P    820
#define K2P    656
#define K3P    525
#define BN_EPS 1e-5f

typedef __attribute__((ext_vector_type(16))) _Float16 v16h;
typedef __attribute__((ext_vector_type(8)))  float    v8f;

// ---------------- elementwise helpers ----------------
__global__ void copy_f32(float* __restrict__ dst, const float* __restrict__ src, int n) {
  int i = blockIdx.x * blockDim.x + threadIdx.x;
  if (i < n) dst[i] = src[i];
}

__global__ void add_f32(float* __restrict__ out, const float* __restrict__ a,
                        const float* __restrict__ b, int n) {
  int i = blockIdx.x * blockDim.x + threadIdx.x;
  if (i < n) out[i] = a[i] + b[i];
}

// ---------------- GIN aggregation: agg[dst] += h[src] (masked) ----------------
__global__ void scatter_add_edges(const float* __restrict__ h, const int* __restrict__ src,
                                  const int* __restrict__ dst, const int* __restrict__ mask,
                                  float* __restrict__ agg, int nedge, int F) {
  long long i = (long long)blockIdx.x * blockDim.x + threadIdx.x;
  if (i >= (long long)nedge * F) return;
  int e = (int)(i / F);
  int f = (int)(i % F);
  if (mask && !mask[e]) return;
  atomicAdd(&agg[(size_t)dst[e] * F + f], h[(size_t)src[e] * F + f]);
}

// ---------------- tiled WMMA GEMM ----------------
// Z[M, N] = T[M, K] @ W[K, N] + bias,  N = NT*16, K % 32 == 0, M % 16 == 0.
// Block = 8 waves; W is staged f16-TRANSPOSED in LDS ([N][K] halves) so each
// lane's B fragment is one contiguous 32-byte ds read. Each wave owns 16 rows
// and the full N width: one A fragment feeds NT back-to-back WMMAs.
template <int NT>
__global__ void gemm_bias_wmma(const float* __restrict__ T, const float* __restrict__ W,
                               const float* __restrict__ bias, float* __restrict__ Z,
                               int M, int K) {
  constexpr int N = NT * 16;
  extern __shared__ char smem_raw[];
  _Float16* Wlds = (_Float16*)smem_raw;                  // [N][K] halves

  // cooperative stage: coalesced global reads, transposed LDS writes
  for (int idx = threadIdx.x; idx < K * N; idx += blockDim.x) {
    int k = idx / N, n = idx % N;
    Wlds[n * K + k] = (_Float16)W[idx];
  }
  __syncthreads();

  int mt = blockIdx.x * (blockDim.x >> 5) + (threadIdx.x >> 5);
  if (mt * 16 >= M) return;                              // wave-uniform: EXEC all-ones for WMMA
  int lane = threadIdx.x & 31;
  int half = lane >> 4;                                  // 0: lanes 0-15, 1: lanes 16-31
  int l16  = lane & 15;
  int arow = mt * 16 + l16;
  const float* Trow = T + (size_t)arow * K;

  v8f acc[NT];
#pragma unroll
  for (int t = 0; t < NT; ++t) acc[t] = (v8f){};

  for (int k0 = 0; k0 < K; k0 += 32) {
    // A fragment: contiguous k order, element e <-> K = k0 + half*16 + e (same as B). Rows are 32B-aligned.
    const float4* ap0 = (const float4*)(Trow + k0 + half * 16);       // K = k0+half*16 ..+7
    const float4* ap1 = (const float4*)(Trow + k0 + half * 16 + 8);   // K = k0+half*16+8 ..+15
    __builtin_prefetch((const float*)ap0 + 32, 0, 1);
    float4 a0 = ap0[0], a1 = ap0[1], a2 = ap1[0], a3 = ap1[1];
    v16h a;
    a[0] = (_Float16)a0.x;  a[1] = (_Float16)a0.y;  a[2]  = (_Float16)a0.z;  a[3]  = (_Float16)a0.w;
    a[4] = (_Float16)a1.x;  a[5] = (_Float16)a1.y;  a[6]  = (_Float16)a1.z;  a[7]  = (_Float16)a1.w;
    a[8] = (_Float16)a2.x;  a[9] = (_Float16)a2.y;  a[10] = (_Float16)a2.z;  a[11] = (_Float16)a2.w;
    a[12] = (_Float16)a3.x; a[13] = (_Float16)a3.y; a[14] = (_Float16)a3.z;  a[15] = (_Float16)a3.w;

#pragma unroll
    for (int t = 0; t < NT; ++t) {
      // B fragment: element e <-> K = k0 + half*16 + e, col = t*16 + l16.
      // Transposed LDS layout makes this contiguous and 32B-aligned.
      v16h b = *(const v16h*)&Wlds[(t * 16 + l16) * K + k0 + half * 16];
      acc[t] = __builtin_amdgcn_wmma_f32_16x16x32_f16(false, a, false, b,
                                                      (short)0, acc[t], false, false);
    }
  }

  // C/D f32 layout: reg r -> row (r + 8*half), col l16 within tile
#pragma unroll
  for (int t = 0; t < NT; ++t) {
    float bv = bias[t * 16 + l16];
#pragma unroll
    for (int r = 0; r < 8; ++r) {
      int m = r + half * 8;
      Z[(size_t)(mt * 16 + m) * N + t * 16 + l16] = acc[t][r] + bv;
    }
  }
}

// ---------------- BatchNorm (training-mode batch stats) ----------------
__global__ void bn_stats(const float* __restrict__ Z, int M, int N,
                         float* __restrict__ mean, float* __restrict__ var) {
  int col = blockIdx.x;                     // one block per feature column
  float s = 0.f, ss = 0.f;
  for (int r = threadIdx.x; r < M; r += blockDim.x) {
    float v = Z[(size_t)r * N + col];
    s += v; ss += v * v;
  }
  __shared__ float sh[256], sh2[256];
  sh[threadIdx.x] = s; sh2[threadIdx.x] = ss;
  __syncthreads();
  for (int st = blockDim.x >> 1; st > 0; st >>= 1) {
    if (threadIdx.x < st) { sh[threadIdx.x] += sh[threadIdx.x + st];
                            sh2[threadIdx.x] += sh2[threadIdx.x + st]; }
    __syncthreads();
  }
  if (threadIdx.x == 0) {
    float m = sh[0] / (float)M;
    mean[col] = m;
    var[col]  = sh2[0] / (float)M - m * m;
  }
}

__global__ void bn_apply_relu(float* __restrict__ Z, int total, int N,
                              const float* __restrict__ mean, const float* __restrict__ var,
                              const float* __restrict__ g, const float* __restrict__ be) {
  int i = blockIdx.x * blockDim.x + threadIdx.x;
  if (i >= total) return;
  int c = i % N;
  float y = (Z[i] - mean[c]) * rsqrtf(var[c] + BN_EPS) * g[c] + be[c];
  Z[i] = fmaxf(y, 0.f);
}

// ---------------- TopK pooling ----------------
__global__ void score_kernel(const float* __restrict__ h, const float* __restrict__ w,
                             float* __restrict__ score, int Nn, int F) {
  __shared__ float wsh[128];
  __shared__ float inv_nrm;
  for (int j = threadIdx.x; j < F; j += blockDim.x) wsh[j] = w[j];
  __syncthreads();
  if (threadIdx.x == 0) {
    float s = 0.f;
    for (int j = 0; j < F; ++j) s += wsh[j] * wsh[j];
    inv_nrm = rsqrtf(s);
  }
  __syncthreads();
  int i = blockIdx.x * blockDim.x + threadIdx.x;
  if (i < Nn) {
    const float* row = h + (size_t)i * F;
    float d = 0.f;
    for (int j = 0; j < F; ++j) d += row[j] * wsh[j];
    score[i] = d * inv_nrm;
  }
}

// One block per graph: rank-select top-k (tie-break by index), compact relabel.
__global__ void topk_kernel(const float* __restrict__ score, int n, int k,
                            int* __restrict__ newid, int* __restrict__ perm,
                            float* __restrict__ gate) {
  int g = blockIdx.x;
  const float* s = score + (size_t)g * n;
  extern __shared__ char smem[];
  float* ss = (float*)smem;
  int*  sel = (int*)(ss + n);
  for (int i = threadIdx.x; i < n; i += blockDim.x) ss[i] = s[i];
  __syncthreads();
  for (int i = threadIdx.x; i < n; i += blockDim.x) {
    float si = ss[i];
    int rank = 0;
    for (int j = 0; j < n; ++j) {
      float sj = ss[j];
      rank += (sj > si) || (sj == si && j < i);
    }
    sel[i] = (rank < k) ? 1 : 0;
  }
  __syncthreads();
  for (int i = threadIdx.x; i < n; i += blockDim.x) {
    if (sel[i]) {
      int pos = 0;
      for (int j = 0; j < i; ++j) pos += sel[j];   // stable compaction (order-invariant downstream)
      int row = g * k + pos;
      newid[g * n + i] = row;
      perm[row] = g * n + i;
      gate[row] = tanhf(ss[i]);
    } else {
      newid[g * n + i] = -1;
    }
  }
}

__global__ void gather_gate(const float* __restrict__ h, const int* __restrict__ perm,
                            const float* __restrict__ gate, float* __restrict__ out,
                            int rows, int F) {
  long long i = (long long)blockIdx.x * blockDim.x + threadIdx.x;
  if (i >= (long long)rows * F) return;
  int r = (int)(i / F);
  int f = (int)(i % F);
  out[i] = h[(size_t)perm[r] * F + f] * gate[r];
}

__global__ void edge_remap(const int* __restrict__ src, const int* __restrict__ dst,
                           const int* __restrict__ mask, const int* __restrict__ newid,
                           int* __restrict__ nsrc, int* __restrict__ ndst,
                           int* __restrict__ nmask, int nedge) {
  int e = blockIdx.x * blockDim.x + threadIdx.x;
  if (e >= nedge) return;
  int ok = mask ? mask[e] : 1;
  int ns = ok ? newid[src[e]] : -1;
  int nd = ok ? newid[dst[e]] : -1;
  int m  = (ok && ns >= 0 && nd >= 0) ? 1 : 0;
  nsrc[e]  = m ? ns : 0;
  ndst[e]  = m ? nd : 0;
  nmask[e] = m;
}

// ---------------- readout: [max || mean] per graph ----------------
__global__ void readout_kernel(const float* __restrict__ h, int k, int F,
                               float* __restrict__ out) {
  int g = blockIdx.x;
  for (int f = threadIdx.x; f < F; f += blockDim.x) {
    const float* base = h + (size_t)g * k * F + f;
    float mx = -INFINITY, sm = 0.f;
    for (int r = 0; r < k; ++r) {
      float v = base[(size_t)r * F];
      mx = fmaxf(mx, v);
      sm += v;
    }
    out[(size_t)g * 2 * F + f]     = mx;
    out[(size_t)g * 2 * F + F + f] = sm / (float)k;
  }
}

// ---------------- final: logits = (d2+x1)@W + b ; log_softmax ----------------
__global__ void final_kernel(const float* __restrict__ d2, const float* __restrict__ x1,
                             const float* __restrict__ W, const float* __restrict__ b,
                             float* __restrict__ out) {
  int row = blockIdx.x;
  __shared__ float inbuf[64];
  __shared__ float logits[10];
  for (int j = threadIdx.x; j < 64; j += blockDim.x)
    inbuf[j] = d2[row * 64 + j] + x1[row * 64 + j];
  __syncthreads();
  for (int c = threadIdx.x; c < 10; c += blockDim.x) {
    float s = b[c];
    for (int j = 0; j < 64; ++j) s += inbuf[j] * W[j * 10 + c];
    logits[c] = s;
  }
  __syncthreads();
  if (threadIdx.x == 0) {
    float mx = logits[0];
    for (int c = 1; c < 10; ++c) mx = fmaxf(mx, logits[c]);
    float se = 0.f;
    for (int c = 0; c < 10; ++c) se += expf(logits[c] - mx);
    float lse = mx + logf(se);
    for (int c = 0; c < 10; ++c) out[row * 10 + c] = logits[c] - lse;
  }
}

// ---------------- host-side helpers ----------------
struct MlpP { const float *W1,*b1,*g1,*be1,*W2,*b2,*g2,*be2; };

static inline int cdiv(long long a, int b) { return (int)((a + b - 1) / b); }

static inline void launch_gemm(const float* A, const float* W, const float* bias, float* Z,
                               int M, int K, int N, hipStream_t s) {
  int blocks = cdiv(M / 16, 8);             // 8 waves (16 rows each) per block
  size_t lds = (size_t)K * N * sizeof(_Float16);
  switch (N) {
    case 32:  gemm_bias_wmma<2><<<blocks, 256, lds, s>>>(A, W, bias, Z, M, K); break;
    case 64:  gemm_bias_wmma<4><<<blocks, 256, lds, s>>>(A, W, bias, Z, M, K); break;
    case 128: gemm_bias_wmma<8><<<blocks, 256, lds, s>>>(A, W, bias, Z, M, K); break;
    default:  break;                        // all layer widths are 32/64/128
  }
}

static inline void run_bn_relu(float* Z, int M, int N, const float* g, const float* be,
                               float* mean, float* var, hipStream_t s) {
  bn_stats<<<N, 256, 0, s>>>(Z, M, N, mean, var);
  int tot = M * N;
  bn_apply_relu<<<(tot + 255) / 256, 256, 0, s>>>(Z, tot, N, mean, var, g, be);
}

static inline void run_mlp(const float* Tin, int M, int Fin, int Fh, const MlpP& p,
                           float* z1, float* z2, float* mean, float* var, hipStream_t s) {
  launch_gemm(Tin, p.W1, p.b1, z1, M, Fin, Fh, s);
  run_bn_relu(z1, M, Fh, p.g1, p.be1, mean, var, s);
  launch_gemm(z1, p.W2, p.b2, z2, M, Fh, Fh, s);
  run_bn_relu(z2, M, Fh, p.g2, p.be2, mean, var, s);
}

extern "C" void kernel_launch(void* const* d_in, const int* in_sizes, int n_in,
                              void* d_out, int out_size, void* d_ws, size_t ws_size,
                              hipStream_t stream) {
  (void)in_sizes; (void)n_in; (void)out_size; (void)ws_size;

  // ---- inputs (setup_inputs dict order, params flattened in insertion order)
  const float* x    = (const float*)d_in[0];          // [65536, 64]
  const int*   src0 = (const int*)d_in[1];            // [2, E]: row0 src
  const int*   dst0 = src0 + E_TOT;                   //         row1 dst
  int p = 3;                                          // d_in[2] = batch (unused)
  auto getMlp = [&](void)->MlpP {
    MlpP m;
    m.W1 = (const float*)d_in[p++]; m.b1 = (const float*)d_in[p++];
    m.g1 = (const float*)d_in[p++]; m.be1 = (const float*)d_in[p++];
    m.W2 = (const float*)d_in[p++]; m.b2 = (const float*)d_in[p++];
    m.g2 = (const float*)d_in[p++]; m.be2 = (const float*)d_in[p++];
    return m;
  };
  MlpP conv1 = getMlp();  const float* p1 = (const float*)d_in[p++];
  MlpP conv2 = getMlp();  const float* p2 = (const float*)d_in[p++];
  MlpP conv3 = getMlp();  const float* p3 = (const float*)d_in[p++];
  MlpP dec3  = getMlp();
  MlpP dec2  = getMlp();
  const float* dec1W = (const float*)d_in[p++];
  const float* dec1b = (const float*)d_in[p++];

  // ---- workspace carve
  char* wp = (char*)d_ws;
  auto take = [&](size_t bytes)->void* {
    void* r = (void*)wp;
    wp += (bytes + 255) & ~(size_t)255;
    return r;
  };
  const size_t PBUF = 6u * 1024u * 1024u;             // floats per ping-pong buffer
  float* P0    = (float*)take(PBUF * 4);              // t / z2 / current h
  float* P1    = (float*)take(PBUF * 4);              // z1
  float* P2    = (float*)take(PBUF * 4);              // pooled h
  float* score = (float*)take(65536 * 4);
  float* gate  = (float*)take(65536 * 4);
  int*   newid = (int*)  take(65536 * 4);
  int*   perm  = (int*)  take(65536 * 4);
  int*   es0   = (int*)  take((size_t)E_TOT * 4);
  int*   ed0   = (int*)  take((size_t)E_TOT * 4);
  int*   em0   = (int*)  take((size_t)E_TOT * 4);
  int*   es1   = (int*)  take((size_t)E_TOT * 4);
  int*   ed1   = (int*)  take((size_t)E_TOT * 4);
  int*   em1   = (int*)  take((size_t)E_TOT * 4);
  float* mean  = (float*)take(256 * 4);
  float* var   = (float*)take(256 * 4);
  float* x1    = (float*)take(64 * 64 * 4);
  float* x2    = (float*)take(64 * 128 * 4);
  float* x3    = (float*)take(64 * 256 * 4);
  float* dz1   = (float*)take(64 * 128 * 4);
  float* dz2   = (float*)take(64 * 128 * 4);
  float* in2   = (float*)take(64 * 128 * 4);

  const int N0 = B_GR * NPER;                         // 65536
  const int N1 = B_GR * K1P;                          // 52480
  const int N2 = B_GR * K2P;                          // 41984
  const int N3 = B_GR * K3P;                          // 33600
  float* out = (float*)d_out;

  // ================= conv1 (GIN: t = x + sum_msg; MLP 64->32->32) =================
  copy_f32<<<cdiv((long long)N0 * 64, 256), 256, 0, stream>>>(P0, x, N0 * 64);
  scatter_add_edges<<<cdiv((long long)E_TOT * 64, 256), 256, 0, stream>>>(
      x, src0, dst0, nullptr, P0, E_TOT, 64);
  run_mlp(P0, N0, 64, 32, conv1, P1, P0, mean, var, stream);   // h1 in P0 [N0,32]

  // pool1 -> K1 per graph
  score_kernel<<<cdiv(N0, 256), 256, 0, stream>>>(P0, p1, score, N0, 32);
  topk_kernel<<<B_GR, 256, NPER * 8, stream>>>(score, NPER, K1P, newid, perm, gate);
  gather_gate<<<cdiv((long long)N1 * 32, 256), 256, 0, stream>>>(P0, perm, gate, P2, N1, 32);
  edge_remap<<<cdiv(E_TOT, 256), 256, 0, stream>>>(src0, dst0, nullptr, newid,
                                                   es0, ed0, em0, E_TOT);
  readout_kernel<<<B_GR, 128, 0, stream>>>(P2, K1P, 32, x1);   // [64, 64]

  // ================= conv2 (MLP 32->64->64) =================
  copy_f32<<<cdiv((long long)N1 * 32, 256), 256, 0, stream>>>(P0, P2, N1 * 32);
  scatter_add_edges<<<cdiv((long long)E_TOT * 32, 256), 256, 0, stream>>>(
      P2, es0, ed0, em0, P0, E_TOT, 32);
  run_mlp(P0, N1, 32, 64, conv2, P1, P0, mean, var, stream);   // h2 in P0 [N1,64]

  // pool2 -> K2 per graph
  score_kernel<<<cdiv(N1, 256), 256, 0, stream>>>(P0, p2, score, N1, 64);
  topk_kernel<<<B_GR, 256, K1P * 8, stream>>>(score, K1P, K2P, newid, perm, gate);
  gather_gate<<<cdiv((long long)N2 * 64, 256), 256, 0, stream>>>(P0, perm, gate, P2, N2, 64);
  edge_remap<<<cdiv(E_TOT, 256), 256, 0, stream>>>(es0, ed0, em0, newid,
                                                   es1, ed1, em1, E_TOT);
  readout_kernel<<<B_GR, 128, 0, stream>>>(P2, K2P, 64, x2);   // [64, 128]

  // ================= conv3 (MLP 64->128->128) =================
  copy_f32<<<cdiv((long long)N2 * 64, 256), 256, 0, stream>>>(P0, P2, N2 * 64);
  scatter_add_edges<<<cdiv((long long)E_TOT * 64, 256), 256, 0, stream>>>(
      P2, es1, ed1, em1, P0, E_TOT, 64);
  run_mlp(P0, N2, 64, 128, conv3, P1, P0, mean, var, stream);  // h3 in P0 [N2,128]

  // pool3 -> K3 per graph (edges after pool3 are unused by the reference output)
  score_kernel<<<cdiv(N2, 256), 256, 0, stream>>>(P0, p3, score, N2, 128);
  topk_kernel<<<B_GR, 256, K2P * 8, stream>>>(score, K2P, K3P, newid, perm, gate);
  gather_gate<<<cdiv((long long)N3 * 128, 256), 256, 0, stream>>>(P0, perm, gate, P2, N3, 128);
  readout_kernel<<<B_GR, 128, 0, stream>>>(P2, K3P, 128, x3);  // [64, 256]

  // ================= decoder =================
  // d3 = relu(mlp(x3, dec3)): 256->128->128
  launch_gemm(x3, dec3.W1, dec3.b1, dz1, 64, 256, 128, stream);
  run_bn_relu(dz1, 64, 128, dec3.g1, dec3.be1, mean, var, stream);
  launch_gemm(dz1, dec3.W2, dec3.b2, dz2, 64, 128, 128, stream);
  run_bn_relu(dz2, 64, 128, dec3.g2, dec3.be2, mean, var, stream);
  // d2 = relu(mlp(d3 + x2, dec2)): 128->64->64
  add_f32<<<cdiv(64 * 128, 256), 256, 0, stream>>>(in2, dz2, x2, 64 * 128);
  launch_gemm(in2, dec2.W1, dec2.b1, dz1, 64, 128, 64, stream);
  run_bn_relu(dz1, 64, 64, dec2.g1, dec2.be1, mean, var, stream);
  launch_gemm(dz1, dec2.W2, dec2.b2, dz2, 64, 64, 64, stream);
  run_bn_relu(dz2, 64, 64, dec2.g2, dec2.be2, mean, var, stream);
  // logits + log_softmax
  final_kernel<<<B_GR, 32, 0, stream>>>(dz2, x1, dec1W, dec1b, out);
}
